// VectorQuantizer_65996467470952
// MI455X (gfx1250) — matrix operand order repaired and checked by reading, not compiled, not run on hardware
//
#include <hip/hip_runtime.h>
#include <hip/hip_bf16.h>

typedef __attribute__((ext_vector_type(16))) __bf16 v16bf;
typedef __attribute__((ext_vector_type(8)))  float  v8f;

#define N_Z   16384   // 16 * 32 * 32 flattened z vectors
#define N_E   8192    // codebook entries
#define DIM   32      // embedding dim == WMMA K for bf16
#define HW    1024    // 32*32
#define CHW   32768   // 32*1024
#define OUT_ELEMS 524288  // 16*32*32*32

__device__ __forceinline__ unsigned short f2bf(float f) {
    unsigned u = __float_as_uint(f);
    unsigned r = (u + 0x7FFFu + ((u >> 16) & 1u)) >> 16;  // round-to-nearest-even
    return (unsigned short)r;
}

// Prep: codebook -> bf16 + per-row squared norm (fp32)
__global__ __launch_bounds__(256) void vq_prep_cb(const float* __restrict__ cb,
                                                  unsigned short* __restrict__ cb16,
                                                  float* __restrict__ en) {
    int r = blockIdx.x * blockDim.x + threadIdx.x;
    if (r >= N_E) return;
    float s = 0.0f;
    #pragma unroll
    for (int k = 0; k < DIM; ++k) {
        float v = cb[r * DIM + k];
        s = __builtin_fmaf(v, v, s);
        cb16[r * DIM + k] = f2bf(v);
    }
    en[r] = s;
}

// Prep: z [b,c,h,w] fp32 -> zf [N, 32] bf16 (transpose to channel-last + convert)
__global__ __launch_bounds__(256) void vq_prep_z(const float* __restrict__ z,
                                                 unsigned short* __restrict__ zf16) {
    int t = blockIdx.x * blockDim.x + threadIdx.x;     // 0 .. N_Z*DIM-1
    int n = t >> 5;
    int c = t & 31;
    int b = n >> 10;
    int hw = n & 1023;
    zf16[t] = f2bf(z[b * CHW + c * HW + hw]);
}

// Main: per wave, 16 z-rows vs all 8192 codes via v_wmma_f32_16x16x32_bf16.
// Two-stage software pipeline: issue WMMA for chunk t, consume chunk t-2, so a
// full iteration separates each WMMA from its consumer (no hazard NOPs).
__global__ __launch_bounds__(256) void vq_main(const float* __restrict__ z,
                                               const float* __restrict__ cb,
                                               const unsigned short* __restrict__ zf16,
                                               const unsigned short* __restrict__ cb16,
                                               const float* __restrict__ en,
                                               float* __restrict__ out,
                                               int*   __restrict__ idxOut,
                                               float* __restrict__ partial) {
    const int lane  = threadIdx.x & 31;
    const int wave  = threadIdx.x >> 5;
    const int wgid  = blockIdx.x * 8 + wave;    // 0..1023
    const int rowBase = wgid * 16;              // 16 z-rows per wave
    const int khalf = lane >> 4;                // lane half selects K sub-range

    // ---- A fragment (16x32 bf16): lane l<16 -> row rowBase+l, K0-15
    //                               lane l>=16 -> row rowBase+(l-16), K16-31
    union { v16bf v; uint4 u[2]; } A;
    const uint4* zrow = (const uint4*)(zf16 + (rowBase + (lane & 15)) * DIM); // 4x16B per row
    A.u[0] = zrow[2 * khalf];
    A.u[1] = zrow[2 * khalf + 1];

    float best[8];
    int   bidx[8];
    #pragma unroll
    for (int j = 0; j < 8; ++j) { best[j] = 3.4e38f; bidx[j] = 0; }

    const uint4* cbv = (const uint4*)cb16;
    const v8f zeroC = {};

    // ---- Pipeline prologue: chunks 0 and 1 in flight
    union { v16bf v; uint4 u[2]; } B;
    int   r0 = (lane & 15);
    {
        const uint4* p = cbv + r0 * 4 + khalf * 2;
        B.u[0] = p[0];
        B.u[1] = p[1];
    }
    float e0 = en[r0];
    v8f c0 = __builtin_amdgcn_wmma_f32_16x16x32_bf16(
                 false, A.v, false, B.v, (short)0, zeroC, false, false);

    int   r1 = 16 + (lane & 15);
    {
        const uint4* p = cbv + r1 * 4 + khalf * 2;
        B.u[0] = p[0];
        B.u[1] = p[1];
    }
    float e1 = en[r1];
    v8f c1 = __builtin_amdgcn_wmma_f32_16x16x32_bf16(
                 false, A.v, false, B.v, (short)0, zeroC, false, false);

    #pragma unroll 2
    for (int t = 2; t < N_E / 16; ++t) {
        // ---- B fragment (32x16 bf16): column N = lane&15 -> code row 16t+N
        //      lanes 0-15 hold K0-15, lanes 16-31 hold K16-31
        int crow = (t << 4) + (lane & 15);
        const uint4* p = cbv + crow * 4 + khalf * 2;
        B.u[0] = p[0];
        B.u[1] = p[1];
        float e2 = en[crow];

        v8f cnew = __builtin_amdgcn_wmma_f32_16x16x32_bf16(
                       false, A.v, false, B.v, (short)0, zeroC, false, false);

        // Consume chunk t-2 (c0): a whole iteration separates it from its WMMA.
        // score = ||e||^2 - 2 * (z . e); ||z||^2 constant per row -> irrelevant.
        #pragma unroll
        for (int j = 0; j < 8; ++j) {
            float s = __builtin_fmaf(-2.0f, c0[j], e0);
            bool lt = s < best[j];               // strict < keeps earliest code
            best[j] = lt ? s : best[j];
            bidx[j] = lt ? r0 : bidx[j];
        }

        c0 = c1; e0 = e1; r0 = r1;
        c1 = cnew; e1 = e2; r1 = crow;
    }

    // ---- Pipeline epilogue: consume the last two chunks
    #pragma unroll
    for (int j = 0; j < 8; ++j) {
        float s = __builtin_fmaf(-2.0f, c0[j], e0);
        bool lt = s < best[j];
        best[j] = lt ? s : best[j];
        bidx[j] = lt ? r0 : bidx[j];
    }
    #pragma unroll
    for (int j = 0; j < 8; ++j) {
        float s = __builtin_fmaf(-2.0f, c1[j], e1);
        bool lt = s < best[j];
        best[j] = lt ? s : best[j];
        bidx[j] = lt ? r1 : bidx[j];
    }

    // ---- Cross-lane argmin within each 16-lane group (rows j and j+8 live in
    //      lanes 0-15 / 16-31 respectively). Lexicographic (score, idx) min.
    #pragma unroll
    for (int j = 0; j < 8; ++j) {
        float b = best[j]; int bi = bidx[j];
        #pragma unroll
        for (int off = 8; off >= 1; off >>= 1) {
            float ob = __shfl_xor(b, off, 32);
            int   oi = __shfl_xor(bi, off, 32);
            if (ob < b || (ob == b && oi < bi)) { b = ob; bi = oi; }
        }
        best[j] = b; bidx[j] = bi;
    }

    // ---- Write indices (row M=j in lanes 0-15, row M=8+j in lanes 16-31)
    #pragma unroll
    for (int j = 0; j < 8; ++j) {
        if (lane == j)      idxOut[rowBase + j]     = bidx[j];
        if (lane == 16 + j) idxOut[rowBase + 8 + j] = bidx[j];
    }

    // ---- Gather exact fp32 codebook rows, scatter to bchw out, accumulate loss
    float acc = 0.0f;
    #pragma unroll
    for (int r = 0; r < 16; ++r) {
        int code = __shfl(bidx[r & 7], (r < 8) ? 0 : 16, 32);
        int n  = rowBase + r;
        int b  = n >> 10;
        int hw = n & 1023;
        float q  = cb[code * DIM + lane];            // channel = lane
        long off = (long)b * CHW + (long)lane * HW + hw;
        float zv = z[off];
        out[off] = q;                                // straight-through forward == z_q
        float d = q - zv;
        acc = __builtin_fmaf(d, d, acc);
    }
    #pragma unroll
    for (int off = 16; off >= 1; off >>= 1) acc += __shfl_xor(acc, off, 32);
    if (lane == 0) partial[wgid] = acc;
}

// Deterministic fixed-order reduction of 1024 partial sums -> scaled loss
__global__ void vq_loss(const float* __restrict__ partial, float* __restrict__ lossOut) {
    int lane = threadIdx.x;  // single wave of 32
    float s = 0.0f;
    for (int i = 0; i < 32; ++i) s += partial[i * 32 + lane];
    #pragma unroll
    for (int off = 16; off >= 1; off >>= 1) s += __shfl_xor(s, off, 32);
    if (lane == 0) *lossOut = s * (1.25f / (float)OUT_ELEMS);  // (1+beta)*mean
}

extern "C" void kernel_launch(void* const* d_in, const int* in_sizes, int n_in,
                              void* d_out, int out_size, void* d_ws, size_t ws_size,
                              hipStream_t stream) {
    const float* z  = (const float*)d_in[0];   // [16,32,32,32] fp32
    const float* cb = (const float*)d_in[1];   // [8192,32] fp32

    // workspace layout
    unsigned short* zf16 = (unsigned short*)d_ws;                          // 1 MB
    unsigned short* cb16 = (unsigned short*)((char*)d_ws + (1u << 20));    // 512 KB
    float* en      = (float*)((char*)d_ws + (1u << 20) + (512u << 10));    // 32 KB
    float* partial = (float*)((char*)d_ws + (1u << 20) + (512u << 10) + (32u << 10)); // 4 KB

    // output layout: out[524288] f32 | loss[1] f32 | idx[16384] int32
    float* out     = (float*)d_out;
    float* lossOut = out + OUT_ELEMS;
    int*   idxOut  = (int*)(out + OUT_ELEMS + 1);

    vq_prep_cb<<<N_E / 256, 256, 0, stream>>>(cb, cb16, en);
    vq_prep_z<<<(N_Z * DIM) / 256, 256, 0, stream>>>(z, zf16);
    vq_main<<<N_Z / 128, 256, 0, stream>>>(z, cb, zf16, cb16, en, out, idxOut, partial);
    vq_loss<<<1, 32, 0, stream>>>(partial, lossOut);
}
